// FeatureAGCNStyle_30597347017061
// MI455X (gfx1250) — matrix operand order repaired and checked by reading, not compiled, not run on hardware
//
#include <hip/hip_runtime.h>

#define NJ 42
#define NS 2            // samples per block
#define THREADS 256     // 8 wave32

typedef __attribute__((ext_vector_type(16))) __bf16 v16bf;
typedef __attribute__((ext_vector_type(2)))  __bf16 v2bf;
typedef __attribute__((ext_vector_type(8)))  float  v8f;

// hand skeleton edges (21 joints per hand)
static __device__ const unsigned char EDGE_A[23] = {0,1,2,3,0,5,6,7,0,9,10,11,0,13,14,15,0,17,18,19,5,9,13};
static __device__ const unsigned char EDGE_B[23] = {1,2,3,4,5,6,7,8,9,10,11,12,13,14,15,16,17,18,19,20,9,13,17};

#if defined(__has_builtin)
#if __has_builtin(__builtin_amdgcn_cvt_pk_bf16_f32)
#define HAVE_PK_BF16 1
#endif
#endif

static __device__ __forceinline__ v2bf pk2(float a, float b) {
#ifdef HAVE_PK_BF16
    return __builtin_amdgcn_cvt_pk_bf16_f32(a, b);
#else
    v2bf r; r[0] = (__bf16)a; r[1] = (__bf16)b; return r;
#endif
}

static __device__ __forceinline__ float base_adj_val(int i, int j) {
    if (i == j) return 1.0f;                                     // + eye
    if ((i == 0 && j == 21) || (i == 21 && j == 0)) return 1.0f; // wrist-wrist link
    if ((i < 21) != (j < 21)) return 0.0f;
    int a = i % 21, b = j % 21;
    #pragma unroll
    for (int e = 0; e < 23; ++e) {
        int ea = EDGE_A[e], eb = EDGE_B[e];
        if ((ea == a && eb == b) || (ea == b && eb == a)) return 1.0f;
    }
    return 0.0f;
}

union TileU {
    v16bf v;
    v2bf  p[8];
    uint4 q[2];
    __bf16 h[16];
};

// ---- WMMA tile loaders (ISA 7.12.2 layouts) -------------------------------
// A (16x32 bf16): lane L<16 -> row L, K = {0..7,16..23}; lane>=16 -> +8.
// Works for LDS or global bf16 source (row-major, K contiguous): 2 x b128.
static __device__ __forceinline__ v16bf ldA16(const __bf16* tile, int ldm, int lrow, int khi) {
    TileU u;
    const __bf16* rp = tile + lrow * ldm + khi * 8;
    u.q[0] = *(const uint4*)(rp);
    u.q[1] = *(const uint4*)(rp + 16);
    return u.v;
}
// A from global fp32 row (rp = &X[row][k0]); four global_load_b128 + pk cvt.
static __device__ __forceinline__ v16bf ldA_g32(const float* rp, int khi) {
    TileU u;
    const float* p = rp + khi * 8;
    float4 f0 = *(const float4*)(p);
    float4 f1 = *(const float4*)(p + 4);
    float4 f2 = *(const float4*)(p + 16);
    float4 f3 = *(const float4*)(p + 20);
    u.p[0] = pk2(f0.x, f0.y); u.p[1] = pk2(f0.z, f0.w);
    u.p[2] = pk2(f1.x, f1.y); u.p[3] = pk2(f1.z, f1.w);
    u.p[4] = pk2(f2.x, f2.y); u.p[5] = pk2(f2.z, f2.w);
    u.p[6] = pk2(f3.x, f3.y); u.p[7] = pk2(f3.z, f3.w);
    return u.v;
}
// B (32x16 bf16): lane L<16 -> col L, K = {0..7,16..23}; lane>=16 -> col L-16, +8.
// cp = column/row base with K contiguous (LDS col-major act or bf16 weight row).
static __device__ __forceinline__ v16bf ldB16(const __bf16* cp, int khi) {
    TileU u;
    const __bf16* p = cp + khi * 8;
    u.q[0] = *(const uint4*)(p);
    u.q[1] = *(const uint4*)(p + 16);
    return u.v;
}

static __device__ __forceinline__ v8f wmma_bf16(v16bf a, v16bf b, v8f c) {
    return __builtin_amdgcn_wmma_f32_16x16x32_bf16(false, a, false, b, (short)0, c, false, false);
}

// Pack 8 floats -> 8 bf16 (one uint4)
static __device__ __forceinline__ uint4 pk8(const float* f) {
    union { uint4 q; v2bf p[4]; } u;
    u.p[0] = pk2(f[0], f[1]); u.p[1] = pk2(f[2], f[3]);
    u.p[2] = pk2(f[4], f[5]); u.p[3] = pk2(f[6], f[7]);
    return u.q;
}

// ============================================================================
// prep: once-per-launch. Block 0 additionally builds softmax(adj_norm + B_l)
// as zero-padded 48x64 bf16; all blocks convert W0/W1/W2 fp32 -> bf16 pairs.
// ws layout (bf16 elements): [adj 2*48*64][W0 64*64][W1 128*64][W2 256*128]
// ============================================================================
__global__ __launch_bounds__(THREADS) void agcn_prep(
    const float* __restrict__ W0, const float* __restrict__ W1, const float* __restrict__ W2,
    const float* __restrict__ B1, const float* __restrict__ B2,
    __bf16* __restrict__ wsAdj, __bf16* __restrict__ wsW0,
    __bf16* __restrict__ wsW1, __bf16* __restrict__ wsW2)
{
    const int tid = threadIdx.x;
    const int gid = blockIdx.x * THREADS + tid;

    // -- weight conversion (grid-wide, pair per thread) --
    if (gid < 64 * 64 / 2)   *(v2bf*)&wsW0[2 * gid] = pk2(W0[2 * gid], W0[2 * gid + 1]);
    if (gid < 128 * 64 / 2)  *(v2bf*)&wsW1[2 * gid] = pk2(W1[2 * gid], W1[2 * gid + 1]);
    if (gid < 256 * 128 / 2) *(v2bf*)&wsW2[2 * gid] = pk2(W2[2 * gid], W2[2 * gid + 1]);

    // -- adjacency build + softmax (block 0 only) --
    if (blockIdx.x == 0) {
        __shared__ float adjS[42 * 44];
        for (int l = 0; l < 2; ++l) {
            const float* Bp = (l == 0) ? B1 : B2;
            __syncthreads();
            for (int idx = tid; idx < 42 * 42; idx += THREADS) {
                int i = idx / 42, j = idx % 42;
                adjS[i * 44 + j] = base_adj_val(i, j);
            }
            __syncthreads();
            if (tid < 42) {                   // normalize row, add B, softmax
                float* row = &adjS[tid * 44];
                float rs = 0.0f;
                for (int j = 0; j < 42; ++j) rs += row[j];
                float inv = 1.0f / rs, mx = -3.0e38f;
                for (int j = 0; j < 42; ++j) { row[j] = row[j] * inv + Bp[tid * 42 + j]; mx = fmaxf(mx, row[j]); }
                float ssum = 0.0f;
                for (int j = 0; j < 42; ++j) { float e = __expf(row[j] - mx); row[j] = e; ssum += e; }
                float is = 1.0f / ssum;
                for (int j = 0; j < 42; ++j) row[j] *= is;
            }
            __syncthreads();
            for (int idx = tid; idx < 48 * 32; idx += THREADS) {   // paired stores
                int i = idx / 32, j2 = (idx % 32) * 2;
                float a = (i < 42 && j2     < 42) ? adjS[i * 44 + j2]     : 0.0f;
                float b = (i < 42 && j2 + 1 < 42) ? adjS[i * 44 + j2 + 1] : 0.0f;
                *(v2bf*)&wsAdj[l * (48 * 64) + i * 64 + j2] = pk2(a, b);
            }
        }
    }
}

// ============================================================================
// fused main kernel: per block, NS samples, whole net in LDS, x read once.
// ============================================================================
__global__ __launch_bounds__(THREADS, 1) void agcn_fused(
    const float* __restrict__ x,
    const float* __restrict__ b0, const float* __restrict__ g0, const float* __restrict__ be0,
    const float* __restrict__ b1, const float* __restrict__ g1, const float* __restrict__ be1,
    const float* __restrict__ b2, const float* __restrict__ g2, const float* __restrict__ be2,
    const float* __restrict__ Wc, const float* __restrict__ bcl,
    const __bf16* __restrict__ wsAdj, const __bf16* __restrict__ wsW0,
    const __bf16* __restrict__ wsW1, const __bf16* __restrict__ wsW2,
    float* __restrict__ out, int N)
{
    // shA* row-major (WMMA A side); sh0/sh1 COLUMN-major (channel-major) so
    // agg B-loads and GEMM epilogue stores are contiguous 16B per lane.
    __shared__ __align__(16) __bf16 sh0 [NS][64 * 64];  // h0^T  [c][v], v=42..63 zero
    __shared__ __align__(16) __bf16 shA1[NS][48 * 64];  // adj1 @ h0   [v][c]
    __shared__ __align__(16) __bf16 sh1 [NS][128 * 64]; // h1^T  [c][v], v=42..63 zero
    __shared__ __align__(16) __bf16 shA2[NS][48 * 128]; // adj2 @ h1   [v][c]
    __shared__ float  feat[NS][256];
    __shared__ float  bnsc[3][42], bnsh[3][42];
    __shared__ float  bias0[64], bias1[128], bias2[256];

    const int tid  = threadIdx.x;
    const int wave = tid >> 5;
    const int lane = tid & 31;
    const int lrow = lane & 15;
    const int khi  = lane >> 4;
    const int n0   = blockIdx.x * NS;

    // ================= step 0: zero padding + params ========================
    {
        const uint4 z4 = {0u, 0u, 0u, 0u};
        uint4* z0 = (uint4*)sh0;
        for (int i = tid; i < NS * 64 * 64 * 2 / 16; i += THREADS) z0[i] = z4;
        uint4* z1 = (uint4*)sh1;
        for (int i = tid; i < NS * 128 * 64 * 2 / 16; i += THREADS) z1[i] = z4;
    }
    const float S = rsqrtf(1.0f + 1e-5f);
    if (tid < 42) {
        bnsc[0][tid] = g0[tid] * S; bnsh[0][tid] = be0[tid];
        bnsc[1][tid] = g1[tid] * S; bnsh[1][tid] = be1[tid];
        bnsc[2][tid] = g2[tid] * S; bnsh[2][tid] = be2[tid];
    }
    if (tid < 64)  bias0[tid] = b0[tid];
    if (tid < 128) bias1[tid] = b1[tid];
    bias2[tid] = b2[tid];

    // prefetch this block's x tile
    {
        int n0c = (n0 < N) ? n0 : (N - 1);
        const float* xblk = x + (size_t)n0c * (NJ * 64);
        for (int i = tid; i < NS * NJ * 2; i += THREADS)   // 128B cachelines
            __builtin_prefetch(xblk + i * 32, 0, 3);
    }
    __syncthreads();

    const int s  = wave >> 2;                 // sample within block
    const int q  = wave & 3;                  // quarter of output columns
    const int sI = n0 + s;
    const float* xbase = x + (size_t)((sI < N) ? sI : (N - 1)) * (NJ * 64);

    // ================= step 1: GEMM0 (x @ W0^T), BN, ReLU -> sh0^T ==========
    {
        v8f acc[3] = {};
        #pragma unroll
        for (int ks = 0; ks < 2; ++ks) {
            int k0 = ks * 32;
            v16bf b = ldB16(wsW0 + (q * 16 + lrow) * 64 + k0, khi);
            #pragma unroll
            for (int mt = 0; mt < 3; ++mt) {
                int vr = mt * 16 + lrow;
                if (vr > NJ - 1) vr = NJ - 1;            // clamp; junk rows dropped
                acc[mt] = wmma_bf16(ldA_g32(xbase + vr * 64 + k0, khi), b, acc[mt]);
            }
        }
        int c = q * 16 + lrow;
        float bcv = bias0[c];
        #pragma unroll
        for (int mt = 0; mt < 3; ++mt) {
            int v0 = mt * 16 + khi * 8;
            float f[8];
            #pragma unroll
            for (int r = 0; r < 8; ++r) {
                int v = v0 + r; int vc = (v < NJ) ? v : (NJ - 1);
                f[r] = fmaxf((acc[mt][r] + bcv) * bnsc[0][vc] + bnsh[0][vc], 0.0f);
            }
            if (v0 + 8 <= NJ) {
                *(uint4*)&sh0[s][c * 64 + v0] = pk8(f);   // one ds_store_b128
            } else {
                #pragma unroll
                for (int r = 0; r < 8; ++r)
                    if (v0 + r < NJ) sh0[s][c * 64 + v0 + r] = (__bf16)f[r];
            }
        }
    }
    __syncthreads();

    // ================= step 2: agg1 = adj1 @ h0 -> shA1 [v][c] ==============
    {
        v8f acc[3] = {};
        #pragma unroll
        for (int ks = 0; ks < 2; ++ks) {
            int k0 = ks * 32;
            v16bf b = ldB16(&sh0[s][(q * 16 + lrow) * 64 + k0], khi);
            #pragma unroll
            for (int mt = 0; mt < 3; ++mt)
                acc[mt] = wmma_bf16(ldA16(wsAdj + (mt * 16) * 64 + k0, 64, lrow, khi), b, acc[mt]);
        }
        int c = q * 16 + lrow;
        #pragma unroll
        for (int mt = 0; mt < 3; ++mt)
            #pragma unroll
            for (int r = 0; r < 8; ++r) {
                int v = mt * 16 + khi * 8 + r;
                shA1[s][v * 64 + c] = (__bf16)acc[mt][r];
            }
    }
    __syncthreads();

    // ================= step 3: GEMM1 (hA1 @ W1^T), BN, ReLU -> sh1^T ========
    {
        v8f acc[3][2] = {};
        #pragma unroll
        for (int ks = 0; ks < 2; ++ks) {
            int k0 = ks * 32;
            v16bf a[3];
            #pragma unroll
            for (int mt = 0; mt < 3; ++mt)
                a[mt] = ldA16(&shA1[s][(mt * 16) * 64 + k0], 64, lrow, khi);
            #pragma unroll
            for (int nt = 0; nt < 2; ++nt) {
                int nc = (q * 2 + nt) * 16;
                v16bf b = ldB16(wsW1 + (nc + lrow) * 64 + k0, khi);
                #pragma unroll
                for (int mt = 0; mt < 3; ++mt)
                    acc[mt][nt] = wmma_bf16(a[mt], b, acc[mt][nt]);
            }
        }
        #pragma unroll
        for (int nt = 0; nt < 2; ++nt) {
            int c = (q * 2 + nt) * 16 + lrow;
            float bcv = bias1[c];
            #pragma unroll
            for (int mt = 0; mt < 3; ++mt) {
                int v0 = mt * 16 + khi * 8;
                float f[8];
                #pragma unroll
                for (int r = 0; r < 8; ++r) {
                    int v = v0 + r; int vc = (v < NJ) ? v : (NJ - 1);
                    f[r] = fmaxf((acc[mt][nt][r] + bcv) * bnsc[1][vc] + bnsh[1][vc], 0.0f);
                }
                if (v0 + 8 <= NJ) {
                    *(uint4*)&sh1[s][c * 64 + v0] = pk8(f);
                } else {
                    #pragma unroll
                    for (int r = 0; r < 8; ++r)
                        if (v0 + r < NJ) sh1[s][c * 64 + v0 + r] = (__bf16)f[r];
                }
            }
        }
    }
    __syncthreads();

    // ================= step 4: agg2 = adj2 @ h1 -> shA2 [v][c] ==============
    {
        v8f acc[3][2] = {};
        #pragma unroll
        for (int ks = 0; ks < 2; ++ks) {
            int k0 = ks * 32;
            v16bf a[3];
            #pragma unroll
            for (int mt = 0; mt < 3; ++mt)
                a[mt] = ldA16(wsAdj + 48 * 64 + (mt * 16) * 64 + k0, 64, lrow, khi);
            #pragma unroll
            for (int nt = 0; nt < 2; ++nt) {
                int nc = (q * 2 + nt) * 16;
                v16bf b = ldB16(&sh1[s][(nc + lrow) * 64 + k0], khi);
                #pragma unroll
                for (int mt = 0; mt < 3; ++mt)
                    acc[mt][nt] = wmma_bf16(a[mt], b, acc[mt][nt]);
            }
        }
        #pragma unroll
        for (int nt = 0; nt < 2; ++nt) {
            int c = (q * 2 + nt) * 16 + lrow;
            #pragma unroll
            for (int mt = 0; mt < 3; ++mt)
                #pragma unroll
                for (int r = 0; r < 8; ++r) {
                    int v = mt * 16 + khi * 8 + r;
                    shA2[s][v * 128 + c] = (__bf16)acc[mt][nt][r];
                }
        }
    }
    __syncthreads();

    // ====== step 5: GEMM2 (hA2 @ W2^T), BN, ReLU, mean over joints =========
    {
        v8f acc[3][4] = {};
        #pragma unroll
        for (int ks = 0; ks < 4; ++ks) {
            int k0 = ks * 32;
            v16bf a[3];
            #pragma unroll
            for (int mt = 0; mt < 3; ++mt)
                a[mt] = ldA16(&shA2[s][(mt * 16) * 128 + k0], 128, lrow, khi);
            #pragma unroll
            for (int nt = 0; nt < 4; ++nt) {
                int nc = (q * 4 + nt) * 16;
                v16bf b = ldB16(wsW2 + (nc + lrow) * 128 + k0, khi);
                #pragma unroll
                for (int mt = 0; mt < 3; ++mt)
                    acc[mt][nt] = wmma_bf16(a[mt], b, acc[mt][nt]);
            }
        }
        #pragma unroll
        for (int nt = 0; nt < 4; ++nt) {
            int c = (q * 4 + nt) * 16 + lrow;
            float bcv = bias2[c];
            float colsum = 0.0f;
            #pragma unroll
            for (int mt = 0; mt < 3; ++mt)
                #pragma unroll
                for (int r = 0; r < 8; ++r) {
                    int v = mt * 16 + khi * 8 + r;
                    if (v < NJ)
                        colsum += fmaxf((acc[mt][nt][r] + bcv) * bnsc[2][v] + bnsh[2][v], 0.0f);
                }
            float other = __shfl_xor(colsum, 16, 32);   // combine the two half-lane rows
            if (khi == 0) feat[s][c] = colsum + other;  // deterministic (one writer)
        }
    }
    __syncthreads();

    // ================= step 6: classifier (mean @ Wc^T + bc) ================
    if (wave < NS * 2) {
        int cs = wave >> 1, k = wave & 1;
        float p = 0.0f;
        for (int i = lane; i < 256; i += 32)
            p += feat[cs][i] * Wc[k * 256 + i];
        #pragma unroll
        for (int off = 16; off > 0; off >>= 1)
            p += __shfl_xor(p, off, 32);
        int si = n0 + cs;
        if (lane == 0 && si < N)
            out[(size_t)si * 2 + k] = p * (1.0f / 42.0f) + bcl[k];
    }
}

extern "C" void kernel_launch(void* const* d_in, const int* in_sizes, int n_in,
                              void* d_out, int out_size, void* d_ws, size_t ws_size,
                              hipStream_t stream) {
    const float* x   = (const float*)d_in[0];
    const float* W0  = (const float*)d_in[1];
    const float* b0  = (const float*)d_in[2];
    const float* g0  = (const float*)d_in[3];
    const float* be0 = (const float*)d_in[4];
    const float* W1  = (const float*)d_in[5];
    const float* b1  = (const float*)d_in[6];
    const float* B1  = (const float*)d_in[7];
    const float* g1  = (const float*)d_in[8];
    const float* be1 = (const float*)d_in[9];
    const float* W2  = (const float*)d_in[10];
    const float* b2  = (const float*)d_in[11];
    const float* B2  = (const float*)d_in[12];
    const float* g2  = (const float*)d_in[13];
    const float* be2 = (const float*)d_in[14];
    const float* Wc  = (const float*)d_in[15];
    const float* bcl = (const float*)d_in[16];
    float* out = (float*)d_out;

    // ws layout (bf16): [adj 2*48*64][W0 64*64][W1 128*64][W2 256*128] = 100 KiB
    __bf16* wsAdj = (__bf16*)d_ws;
    __bf16* wsW0  = wsAdj + 2 * 48 * 64;
    __bf16* wsW1  = wsW0 + 64 * 64;
    __bf16* wsW2  = wsW1 + 128 * 64;

    int N = in_sizes[0] / (NJ * 64);
    int blocks = (N + NS - 1) / NS;

    agcn_prep<<<64, THREADS, 0, stream>>>(W0, W1, W2, B1, B2, wsAdj, wsW0, wsW1, wsW2);
    agcn_fused<<<blocks, THREADS, 0, stream>>>(x, b0, g0, be0, b1, g1, be1, b2, g2, be2,
                                               Wc, bcl, wsAdj, wsW0, wsW1, wsW2, out, N);
    (void)ws_size; (void)n_in; (void)out_size;
}
